// RAN_21406117003953
// MI455X (gfx1250) — compile-verified, matrix operand order checked by reading, not run on hardware
//
#include <hip/hip_runtime.h>

#define S_ 128
#define B_ 32
#define H_ 1024
#define V_ 32000

typedef __attribute__((ext_vector_type(16))) __bf16 v16bf;
typedef __attribute__((ext_vector_type(8)))  float  v8f;
typedef int v4i_ __attribute__((vector_size(16)));   // matches builtin param type

#if defined(__has_builtin)
# if __has_builtin(__builtin_amdgcn_global_load_async_to_lds_b128)
#  define HAVE_ASYNC_LDS 1
# endif
# if __has_builtin(__builtin_amdgcn_s_wait_asynccnt)
#  define HAVE_WAIT_ASYNC 1
# endif
#endif
#ifndef HAVE_ASYNC_LDS
# define HAVE_ASYNC_LDS 0
#endif

union V16BF { v16bf v; uint4 q[2]; };

// float -> bf16 round-to-nearest-even (bit trick; buffers kept as ushort so we
// never need scalar __bf16 arithmetic)
__device__ __forceinline__ unsigned short f2bf(float f) {
    unsigned u = __float_as_uint(f);
    unsigned r = u + 0x7FFFu + ((u >> 16) & 1u);
    return (unsigned short)(r >> 16);
}

// Load one wave-striped 16x32 bf16 operand tile (global or LDS pointer).
// p = base + row(lane&15)*ld + ((lane>>4)*8); lanes 0-15 hold K 0-7 & 16-23,
// lanes 16-31 hold K 8-15 & 24-31 (ISA 7.12.2).
__device__ __forceinline__ v16bf ld_tile(const unsigned short* p) {
    V16BF t;
    t.q[0] = *(const uint4*)(p);
    t.q[1] = *(const uint4*)(p + 16);
    return t.v;
}

__device__ __forceinline__ v8f wmma_bf16(v16bf a, v16bf b, v8f c) {
    return __builtin_amdgcn_wmma_f32_16x16x32_bf16(
        /*neg_a=*/false, a, /*neg_b=*/false, b,
        /*c_mod=*/(short)0, c, /*reuse_a=*/false, /*reuse_b=*/false);
}

__device__ __forceinline__ float sigmoidf(float z) {
    return 1.0f / (1.0f + __expf(-z));
}

__device__ __forceinline__ void wait_async0() {
#if HAVE_ASYNC_LDS
# if defined(HAVE_WAIT_ASYNC)
    __builtin_amdgcn_s_wait_asynccnt(0);
# else
    asm volatile("s_wait_asynccnt 0" ::: "memory");
# endif
#endif
}

// Stage a 128-row x 32-K bf16 panel (8 KB) into LDS: 512 16-byte chunks,
// 2 per thread (blockDim = 256). Async-to-LDS when available (ASYNCcnt path),
// else global_load_b128 + ds_store_b128.
__device__ __forceinline__ void stage_panel(const unsigned short* __restrict__ gbase,
                                            unsigned short* lbase, int k0, int tid) {
    #pragma unroll
    for (int cc = 0; cc < 2; ++cc) {
        int c   = tid + cc * 256;
        int row = c >> 2;
        int q   = c & 3;
        const unsigned short* g = gbase + (size_t)row * H_ + k0 + q * 8;
        unsigned short*       l = lbase + row * 32 + q * 8;
#if HAVE_ASYNC_LDS
        __builtin_amdgcn_global_load_async_to_lds_b128(
            (__attribute__((address_space(1))) v4i_*)g,
            (__attribute__((address_space(3))) v4i_*)l, 0, 0);
#else
        *(uint4*)l = *(const uint4*)g;
#endif
    }
}

// ---------------- conversion / gather kernels ----------------

__global__ void k_f32_to_bf16(const float* __restrict__ src,
                              unsigned short* __restrict__ dst, int n) {
    int i = blockIdx.x * blockDim.x + threadIdx.x;
    if (i < n) dst[i] = f2bf(src[i]);
}

// x[s,b,:] = E[tokens[s,b],:]  (bf16 copy; fp32 x re-gathered in step kernel)
__global__ void k_embed(const int* __restrict__ tokens,
                        const float* __restrict__ E,
                        unsigned short* __restrict__ xbf) {
    int i = blockIdx.x * blockDim.x + threadIdx.x;
    if (i >= S_ * B_ * H_) return;
    int sb = i >> 10;            // H_ = 1024
    int h  = i & (H_ - 1);
    int tok = tokens[sb];
    xbf[i] = f2bf(E[(size_t)tok * H_ + h]);
}

// ---------------- input projections: xi = x@Wix^T, xf = x@Wfx^T ----------------
// M = 4096 (s*B+b), N = 1024 per gate, K = 1024. One 16x16 tile per wave.
__global__ void k_proj(const unsigned short* __restrict__ xbf,
                       const unsigned short* __restrict__ Wix,
                       const unsigned short* __restrict__ Wfx,
                       float* __restrict__ xi, float* __restrict__ xf) {
    int wave = (blockIdx.x * blockDim.x + threadIdx.x) >> 5;   // 0 .. 32767
    int lane = threadIdx.x & 31;
    int mt = wave >> 7;          // 256 M-tiles
    int nt = wave & 127;         // 64 N-tiles for xi, 64 for xf
    const unsigned short* Bw = (nt < 64) ? Wix : Wfx;
    float* C = (nt < 64) ? xi : xf;
    int m0 = mt * 16;
    int n0 = (nt & 63) * 16;

    int r  = lane & 15;
    int kb = (lane >> 4) << 3;
    const unsigned short* ap = xbf + (size_t)(m0 + r) * H_ + kb;
    const unsigned short* bp = Bw  + (size_t)(n0 + r) * H_ + kb;

    v8f acc = {};
    #pragma unroll 4
    for (int k = 0; k < H_; k += 32)
        acc = wmma_bf16(ld_tile(ap + k), ld_tile(bp + k), acc);

    int mo = (lane >> 4) * 8;
    int n  = n0 + (lane & 15);
    #pragma unroll
    for (int rr = 0; rr < 8; ++rr)
        C[(size_t)(m0 + mo + rr) * H_ + n] = acc[rr];
}

// ---------------- recurrent step (launched S_ times, stream-ordered) ----------------
__global__ void k_step(int s,
                       const int* __restrict__ tokens,
                       const float* __restrict__ E,
                       const unsigned short* __restrict__ statebf,
                       const float* __restrict__ statef,
                       const unsigned short* __restrict__ Wic,
                       const unsigned short* __restrict__ Wfc,
                       const float* __restrict__ b_i,
                       const float* __restrict__ b_f,
                       const float* __restrict__ xi,
                       const float* __restrict__ xf,
                       float* __restrict__ outf_s,
                       unsigned short* __restrict__ outbf_s) {
    int wave = (blockIdx.x * blockDim.x + threadIdx.x) >> 5;   // 0 .. 127
    int lane = threadIdx.x & 31;
    int mt = wave & 1;           // 2 M-tiles (B_ = 32)
    int nt = wave >> 1;          // 64 N-tiles
    int m0 = mt * 16, n0 = nt * 16;

    int r  = lane & 15;
    int kb = (lane >> 4) << 3;
    const unsigned short* ap  = statebf + (size_t)(m0 + r) * H_ + kb;
    const unsigned short* bip = Wic + (size_t)(n0 + r) * H_ + kb;
    const unsigned short* bfp = Wfc + (size_t)(n0 + r) * H_ + kb;

    v8f accI = {}, accF = {};
    #pragma unroll 4
    for (int k = 0; k < H_; k += 32) {
        v16bf a = ld_tile(ap + k);
        accI = wmma_bf16(a, ld_tile(bip + k), accI);
        accF = wmma_bf16(a, ld_tile(bfp + k), accF);
    }

    int mo = (lane >> 4) * 8;
    int n  = n0 + (lane & 15);
    float bi = b_i[n], bf = b_f[n];
    #pragma unroll
    for (int rr = 0; rr < 8; ++rr) {
        int m = m0 + mo + rr;                       // batch index
        size_t sbh = ((size_t)s * B_ + m) * H_ + n;
        float it = sigmoidf(accI[rr] + bi + xi[sbh]);
        float ft = sigmoidf(accF[rr] + bf + xf[sbh]);
        int tok = tokens[s * B_ + m];
        float xv = E[(size_t)tok * H_ + n];
        float c  = it * xv + ft * statef[(size_t)m * H_ + n];
        outf_s[(size_t)m * H_ + n]  = c;
        outbf_s[(size_t)m * H_ + n] = f2bf(c);
    }
}

// ---------------- decoder GEMM: decoded = out @ E^T + b_dec ----------------
// M = 4096, N = 32000, K = 1024. Workgroup tile 128M x 128N staged through LDS
// (double-buffered 2 x (8KB A + 8KB B) of the WGP's 320KB). 8 waves, each
// computes 32M x 64N = 8 WMMAs per K-step from ds_load_b128 tiles.
// Cuts L2 read traffic from ~21 GB (no reuse) to ~4.3 GB; output writes
// (524 MB @ 23.3 TB/s) remain the floor.
__global__ void __launch_bounds__(256)
k_dec(const unsigned short* __restrict__ outbf,
      const unsigned short* __restrict__ Ebf,
      const float* __restrict__ b_dec,
      float* __restrict__ out) {
    __shared__ unsigned short As[2][128 * 32];
    __shared__ unsigned short Bs[2][128 * 32];

    int tid  = threadIdx.x;
    int lane = tid & 31;
    int w    = tid >> 5;         // 0..7
    int wm   = w & 3;            // 4 x 32 rows   = 128 M
    int wn   = w >> 2;           // 2 x 64 cols   = 128 N
    int m0   = blockIdx.y * 128;
    int n0   = blockIdx.x * 128;

    const unsigned short* Ag = outbf + (size_t)m0 * H_;
    const unsigned short* Bg = Ebf   + (size_t)n0 * H_;

    v8f acc[2][4] = {};

    // prologue: stage K-step 0 into buffer 0
    stage_panel(Ag, As[0], 0, tid);
    stage_panel(Bg, Bs[0], 0, tid);
    wait_async0();
    __syncthreads();

    int r  = lane & 15;
    int kb = (lane >> 4) << 3;
    int buf = 0;

    for (int k0 = 0; k0 < H_; k0 += 32) {
        int nb = buf ^ 1;
        if (k0 + 32 < H_) {
            stage_panel(Ag, As[nb], k0 + 32, tid);
            stage_panel(Bg, Bs[nb], k0 + 32, tid);
        }

        const unsigned short* Ab = As[buf] + (wm * 32 + r) * 32 + kb;
        const unsigned short* Bb = Bs[buf] + (wn * 64 + r) * 32 + kb;
        v16bf a0 = ld_tile(Ab);
        v16bf a1 = ld_tile(Ab + 16 * 32);
        v16bf b0 = ld_tile(Bb);
        v16bf b1 = ld_tile(Bb + 16 * 32);
        v16bf b2 = ld_tile(Bb + 32 * 32);
        v16bf b3 = ld_tile(Bb + 48 * 32);

        acc[0][0] = wmma_bf16(a0, b0, acc[0][0]);
        acc[0][1] = wmma_bf16(a0, b1, acc[0][1]);
        acc[0][2] = wmma_bf16(a0, b2, acc[0][2]);
        acc[0][3] = wmma_bf16(a0, b3, acc[0][3]);
        acc[1][0] = wmma_bf16(a1, b0, acc[1][0]);
        acc[1][1] = wmma_bf16(a1, b1, acc[1][1]);
        acc[1][2] = wmma_bf16(a1, b2, acc[1][2]);
        acc[1][3] = wmma_bf16(a1, b3, acc[1][3]);

        wait_async0();
        __syncthreads();
        buf = nb;
    }

    int mo = (lane >> 4) * 8;
    int nl = lane & 15;
    #pragma unroll
    for (int i = 0; i < 2; ++i) {
        int gm = m0 + wm * 32 + i * 16 + mo;
        #pragma unroll
        for (int j = 0; j < 4; ++j) {
            int n = n0 + wn * 64 + j * 16 + nl;
            float bd = b_dec[n];
            #pragma unroll
            for (int rr = 0; rr < 8; ++rr)
                out[(size_t)(gm + rr) * V_ + n] = acc[i][j][rr] + bd;
        }
    }
}

__global__ void k_copy_f32(const float* __restrict__ src,
                           float* __restrict__ dst, int n) {
    int i = blockIdx.x * blockDim.x + threadIdx.x;
    if (i < n) dst[i] = src[i];
}

// ---------------- host launch ----------------

extern "C" void kernel_launch(void* const* d_in, const int* in_sizes, int n_in,
                              void* d_out, int out_size, void* d_ws, size_t ws_size,
                              hipStream_t stream) {
    (void)in_sizes; (void)n_in; (void)out_size; (void)ws_size;

    const int*   tokens = (const int*)  d_in[0];   // [S,B]
    const float* hidden = (const float*)d_in[1];   // [1,B,H]
    const float* E      = (const float*)d_in[2];   // [V,H]
    const float* w_ic   = (const float*)d_in[3];
    const float* w_ix   = (const float*)d_in[4];
    const float* w_fc   = (const float*)d_in[5];
    const float* w_fx   = (const float*)d_in[6];
    const float* b_i    = (const float*)d_in[7];
    const float* b_f    = (const float*)d_in[8];
    const float* b_dec  = (const float*)d_in[9];
    float* out = (float*)d_out;                    // decoded [S,B,V] ++ hT [1,B,H]

    // workspace layout (all chunks 256B-aligned by construction)
    char* ws = (char*)d_ws;
    unsigned short* Ebf   = (unsigned short*)(ws);
    unsigned short* Wicb  = (unsigned short*)(ws + (size_t)V_ * H_ * 2);
    unsigned short* Wixb  = Wicb + (size_t)H_ * H_;
    unsigned short* Wfcb  = Wixb + (size_t)H_ * H_;
    unsigned short* Wfxb  = Wfcb + (size_t)H_ * H_;
    unsigned short* xbf   = Wfxb + (size_t)H_ * H_;              // [S,B,H]
    unsigned short* h0bf  = xbf + (size_t)S_ * B_ * H_;          // [B,H]
    float* xi    = (float*)(h0bf + (size_t)B_ * H_);             // [S,B,H]
    float* xf    = xi + (size_t)S_ * B_ * H_;                    // [S,B,H]
    float* outf  = xf + (size_t)S_ * B_ * H_;                    // [S,B,H]
    unsigned short* outbf = (unsigned short*)(outf + (size_t)S_ * B_ * H_);

    // 1) bf16 conversions (E is then L2-resident at 65.5 MB for the decoder)
    k_f32_to_bf16<<<(V_ * H_ + 255) / 256, 256, 0, stream>>>(E,    Ebf,  V_ * H_);
    k_f32_to_bf16<<<(H_ * H_ + 255) / 256, 256, 0, stream>>>(w_ic, Wicb, H_ * H_);
    k_f32_to_bf16<<<(H_ * H_ + 255) / 256, 256, 0, stream>>>(w_ix, Wixb, H_ * H_);
    k_f32_to_bf16<<<(H_ * H_ + 255) / 256, 256, 0, stream>>>(w_fc, Wfcb, H_ * H_);
    k_f32_to_bf16<<<(H_ * H_ + 255) / 256, 256, 0, stream>>>(w_fx, Wfxb, H_ * H_);
    k_f32_to_bf16<<<(B_ * H_ + 255) / 256, 256, 0, stream>>>(hidden, h0bf, B_ * H_);

    // 2) embedding gather (bf16 x for the projection GEMM)
    k_embed<<<(S_ * B_ * H_ + 255) / 256, 256, 0, stream>>>(tokens, E, xbf);

    // 3) input-side projections (no recurrence): 32768 waves
    k_proj<<<4096, 256, 0, stream>>>(xbf, Wixb, Wfxb, xi, xf);

    // 4) sequential scan: 128 stream-ordered fused gate-GEMM steps
    for (int s = 0; s < S_; ++s) {
        const unsigned short* sbf = (s == 0) ? h0bf  : outbf + (size_t)(s - 1) * B_ * H_;
        const float*          sf  = (s == 0) ? hidden : outf + (size_t)(s - 1) * B_ * H_;
        k_step<<<16, 256, 0, stream>>>(s, tokens, E, sbf, sf, Wicb, Wfcb,
                                       b_i, b_f, xi, xf,
                                       outf  + (size_t)s * B_ * H_,
                                       outbf + (size_t)s * B_ * H_);
    }

    // 5) decoder GEMM (dominant): 8000 blocks of 128x128, LDS double-buffered
    k_dec<<<dim3(V_ / 128, (S_ * B_) / 128), 256, 0, stream>>>(outbf, Ebf, b_dec, out);

    // 6) hT = out[S-1] (fp32), appended after decoded
    k_copy_f32<<<(B_ * H_ + 255) / 256, 256, 0, stream>>>(
        outf + (size_t)(S_ - 1) * B_ * H_, out + (size_t)S_ * B_ * V_, B_ * H_);
}
